// HeteroCellNSA_42786464203454
// MI455X (gfx1250) — matrix operand order repaired and checked by reading, not compiled, not run on hardware
//
#include <hip/hip_runtime.h>

// ---------------------------------------------------------------------------
// CDNA5 (gfx1250, wave32) implementation of HeteroCellNSA forward.
// Only the gene path is computed (reaction/metabolite branches are dead code
// w.r.t. the output). All matmuls use v_wmma_f32_16x16x32_bf16.
// GEMM: LDS double-buffered W slabs, async global->LDS staging when available,
// full A-row register preload, branch-free unrolled inner loop.
// ---------------------------------------------------------------------------

typedef unsigned short u16;
typedef __attribute__((ext_vector_type(16))) __bf16 v16bf;
typedef __attribute__((ext_vector_type(8)))  float  v8f;
typedef __attribute__((ext_vector_type(4)))  unsigned int u32x4;
typedef __attribute__((ext_vector_type(4)))  int v4i;

union BFrag { v16bf v; u32x4 q[2]; u16 h[16]; };

#define FLAG_RELU 1
#define FLAG_ACC  2
#define FLAG_TBF  4   // write bf16 output transposed [col][row]

static __device__ __forceinline__ u16 f2bf(float f) {
    unsigned u = __float_as_uint(f);
    u += 0x7fffu + ((u >> 16) & 1u);          // round-to-nearest-even
    return (u16)(u >> 16);
}

static __device__ __forceinline__ v8f wmma_bf16(const BFrag& a, const BFrag& b, v8f c) {
    return __builtin_amdgcn_wmma_f32_16x16x32_bf16(false, a.v, false, b.v,
                                                   (short)0, c, false, false);
}

// ---- async global->LDS staging (CDNA5 GLOBAL_LOAD_ASYNC_TO_LDS_B128) -------
#if defined(__has_builtin)
#if __has_builtin(__builtin_amdgcn_global_load_async_to_lds_b128)
#define HAVE_ASYNC_LDS 1
#endif
#endif

#ifdef HAVE_ASYNC_LDS
static __device__ __forceinline__ void async_cp16(const void* g, void* l) {
    __builtin_amdgcn_global_load_async_to_lds_b128((v4i*)g, (v4i*)l, 0, 0);
}
static __device__ __forceinline__ void wait_async0() {
#if __has_builtin(__builtin_amdgcn_s_wait_asynccnt)
    __builtin_amdgcn_s_wait_asynccnt(0);
#else
    asm volatile("s_wait_asynccnt 0x0" ::: "memory");
#endif
}
#endif

// ---------------------------------------------------------------------------
__global__ void f32_to_bf16_k(const float* __restrict__ x, u16* __restrict__ y, int n) {
    int i = blockIdx.x * 256 + threadIdx.x;
    if (i < n) y[i] = f2bf(x[i]);
}

// transpose + convert: W[K][N] fp32 -> WT[N][K] bf16
__global__ void tconv_k(const float* __restrict__ W, u16* __restrict__ WT, int K, int N) {
    int i = blockIdx.x * 256 + threadIdx.x;
    if (i < K * N) {
        int k = i / N, n = i % N;
        WT[n * K + k] = f2bf(W[i]);
    }
}

// pack bool bytes [rows][cols] into bitmask words [rows][cols/32]
__global__ void pack_mask_k(const unsigned char* __restrict__ m, unsigned* __restrict__ o,
                            int rows, int cols) {
    int words = cols >> 5;
    int i = blockIdx.x * 256 + threadIdx.x;
    if (i < rows * words) {
        int q = i / words, w = i % words;
        const unsigned char* p = m + (size_t)q * cols + (size_t)w * 32;
        unsigned bits = 0;
        #pragma unroll
        for (int b = 0; b < 32; ++b) bits |= (p[b] ? 1u : 0u) << b;
        o[i] = bits;
    }
}

// ---------------------------------------------------------------------------
// GEMM: C[M][N] = A[M][256](bf16) @ WT[N][256](bf16)^T  (+bias, relu, acc)
// TPW = col tiles per wave (N = 128*TPW). 256 threads, 16-row block tile.
// Double-buffered LDS W slabs (32 x N halves each), async staged.
template <int TPW>
__global__ void gemm_bf16_k(const u16* __restrict__ A, const u16* __restrict__ WT,
                            const float* __restrict__ bias,
                            float* __restrict__ C32, u16* __restrict__ Cbf,
                            int M, int flags) {
    constexpr int N = 128 * TPW;
    constexpr int K = 256;
    constexpr int STEPS = K / 32;
    __shared__ u16 lds[2][N * 32];

    const int t = threadIdx.x, w = t >> 5, lane = t & 31;
    const int hl = lane >> 4, ln = lane & 15;
    const int rb = blockIdx.x * 16;

    // preload this wave's entire A row (512B) into registers: 8 fragments
    const u16* pA = A + (size_t)(rb + ln) * K;
    BFrag a[STEPS];
    #pragma unroll
    for (int s = 0; s < STEPS; ++s) {
        a[s].q[0] = *(const u32x4*)(pA + 32 * s + 16 * hl);
        a[s].q[1] = *(const u32x4*)(pA + 32 * s + 16 * hl + 8);
    }

    v8f acc[TPW];
    #pragma unroll
    for (int tt = 0; tt < TPW; ++tt)
        acc[tt] = {0.f,0.f,0.f,0.f,0.f,0.f,0.f,0.f};

    // stage W slab s (WT[:, 32s..32s+31]) into buffer buf
    auto stage = [&](int s, int buf) {
        if (TPW == 2 || t < N) {
            const u16* pw = WT + (size_t)t * K + 32 * s;
            u16* dst = &lds[buf][t * 32];
#ifdef HAVE_ASYNC_LDS
            async_cp16(pw + 0,  dst + 0);
            async_cp16(pw + 8,  dst + 8);
            async_cp16(pw + 16, dst + 16);
            async_cp16(pw + 24, dst + 24);
#else
            u32x4* d = (u32x4*)dst;
            d[0] = *(const u32x4*)(pw + 0);
            d[1] = *(const u32x4*)(pw + 8);
            d[2] = *(const u32x4*)(pw + 16);
            d[3] = *(const u32x4*)(pw + 24);
#endif
        }
    };

    stage(0, 0);
    #pragma unroll
    for (int s = 0; s < STEPS; ++s) {
#ifdef HAVE_ASYNC_LDS
        wait_async0();
#endif
        __syncthreads();                       // slab s ready; prev compute done
        if (s + 1 < STEPS) stage(s + 1, (s + 1) & 1);
        #pragma unroll
        for (int tt = 0; tt < TPW; ++tt) {
            const int ct = w * TPW + tt;
            BFrag b;
            const u16* pb = &lds[s & 1][(ct * 16 + ln) * 32 + 16 * hl];
            b.q[0] = *(const u32x4*)pb;
            b.q[1] = *(const u32x4*)(pb + 8);
            acc[tt] = wmma_bf16(a[s], b, acc[tt]);
        }
    }

    #pragma unroll
    for (int tt = 0; tt < TPW; ++tt) {
        const int ct = w * TPW + tt;
        #pragma unroll
        for (int j = 0; j < 8; ++j) {
            const int row = rb + j + 8 * hl;
            const int col = ct * 16 + ln;
            float v = acc[tt][j];
            if (bias) v += bias[col];
            if (flags & FLAG_ACC) v += C32[(size_t)row * N + col];
            if (flags & FLAG_RELU) v = fmaxf(v, 0.f);
            if (C32) C32[(size_t)row * N + col] = v;
            if (Cbf) {
                u16 bv = f2bf(v);
                if (flags & FLAG_TBF) Cbf[(size_t)col * M + row] = bv;
                else                  Cbf[(size_t)row * N + col] = bv;
            }
        }
    }
}

// ---------------------------------------------------------------------------
// Flash attention, 8 heads, DH=32, H=256. One wave per (head, 16-row q-tile).
// Q,K: bf16 [N][256] (head h in cols h*32..+31). Vt: bf16 [256][Nk] (transposed).
// pm: packed mask [Nq][Nk/32] (MASKED instantiation). Ctx out: bf16 [Nq][256].
template <bool MASKED>
__global__ void flash_k(const u16* __restrict__ Qb, const u16* __restrict__ Kb,
                        const u16* __restrict__ Vt, u16* __restrict__ Ctx,
                        const unsigned* __restrict__ pm, int Nq, int Nk, float scale) {
    __shared__ u16 pbuf[8 * 16 * 32];              // per-wave P bounce (1KB each)
    const int w = threadIdx.x >> 5, lane = threadIdx.x & 31;
    const int hl = lane >> 4, ln = lane & 15;
    const int h = w;
    const int qbase = blockIdx.x * 16;
    u16* pb = pbuf + w * 512;

    BFrag aq;                                      // Q tile 16x32, loaded once
    {
        const u16* pr = Qb + (size_t)(qbase + ln) * 256 + h * 32;
        aq.q[0] = *(const u32x4*)(pr + 16 * hl);
        aq.q[1] = *(const u32x4*)(pr + 16 * hl + 8);
    }

    v8f c0 = {0.f,0.f,0.f,0.f,0.f,0.f,0.f,0.f};
    v8f c1 = {0.f,0.f,0.f,0.f,0.f,0.f,0.f,0.f};
    float m[8], l[8];
    #pragma unroll
    for (int j = 0; j < 8; ++j) { m[j] = -3.0e38f; l[j] = 0.f; }

    const int mwords = Nk >> 5;
    const v8f zf = {0.f,0.f,0.f,0.f,0.f,0.f,0.f,0.f};

    for (int kt = 0; kt < (Nk >> 5); ++kt) {
        const int k0 = kt << 5;
        BFrag bk0, bk1;                            // K^T tiles: K-dim = dh (contig)
        {
            const u16* pr = Kb + (size_t)(k0 + ln) * 256 + h * 32 + 16 * hl;
            bk0.q[0] = *(const u32x4*)pr;  bk0.q[1] = *(const u32x4*)(pr + 8);
        }
        {
            const u16* pr = Kb + (size_t)(k0 + 16 + ln) * 256 + h * 32 + 16 * hl;
            bk1.q[0] = *(const u32x4*)pr;  bk1.q[1] = *(const u32x4*)(pr + 8);
        }
        v8f s0 = wmma_bf16(aq, bk0, zf);
        v8f s1 = wmma_bf16(aq, bk1, zf);

        #pragma unroll
        for (int j = 0; j < 8; ++j) {              // online softmax per q-row
            float a = s0[j] * scale, b = s1[j] * scale;
            if (MASKED) {
                unsigned mw = pm[(size_t)(qbase + j + 8 * hl) * mwords + kt];
                if (!((mw >> ln) & 1u))        a = -1e9f;
                if (!((mw >> (16 + ln)) & 1u)) b = -1e9f;
            }
            float mx = fmaxf(a, b);
            #pragma unroll
            for (int o = 1; o < 16; o <<= 1) mx = fmaxf(mx, __shfl_xor(mx, o, 32));
            float mn = fmaxf(m[j], mx);
            float corr = __expf(m[j] - mn);
            float e0 = __expf(a - mn), e1 = __expf(b - mn);
            float rs = e0 + e1;
            #pragma unroll
            for (int o = 1; o < 16; o <<= 1) rs += __shfl_xor(rs, o, 32);
            l[j] = l[j] * corr + rs;
            m[j] = mn;
            c0[j] *= corr; c1[j] *= corr;
            pb[(j + 8 * hl) * 32 + ln]      = f2bf(e0);   // P -> LDS (bf16)
            pb[(j + 8 * hl) * 32 + 16 + ln] = f2bf(e1);
        }

        BFrag ap, bv0, bv1;                        // P as A-fragment (from LDS)
        ap.q[0] = *(const u32x4*)(pb + ln * 32 + 16 * hl);
        ap.q[1] = *(const u32x4*)(pb + ln * 32 + 16 * hl + 8);
        {
            const u16* pr = Vt + (size_t)(h * 32 + ln) * Nk + k0 + 16 * hl;
            bv0.q[0] = *(const u32x4*)pr;  bv0.q[1] = *(const u32x4*)(pr + 8);
        }
        {
            const u16* pr = Vt + (size_t)(h * 32 + 16 + ln) * Nk + k0 + 16 * hl;
            bv1.q[0] = *(const u32x4*)pr;  bv1.q[1] = *(const u32x4*)(pr + 8);
        }
        c0 = wmma_bf16(ap, bv0, c0);
        c1 = wmma_bf16(ap, bv1, c1);
    }

    #pragma unroll
    for (int j = 0; j < 8; ++j) {
        const float inv = 1.0f / l[j];
        const int row = qbase + j + 8 * hl;
        Ctx[(size_t)row * 256 + h * 32 + ln]      = f2bf(c0[j] * inv);
        Ctx[(size_t)row * 256 + h * 32 + 16 + ln] = f2bf(c1[j] * inv);
    }
}

// ---------------------------------------------------------------------------
// LayerNorm (+optional relu) over rows of 256; one wave per row.
__global__ void ln_act_k(const float* __restrict__ X, const float* __restrict__ g,
                         const float* __restrict__ b, float* __restrict__ Y,
                         u16* __restrict__ Ybf, int relu) {
    const int row = blockIdx.x * 8 + (threadIdx.x >> 5);
    const int lane = threadIdx.x & 31;
    const float* pr = X + (size_t)row * 256;
    float x[8];
    #pragma unroll
    for (int i = 0; i < 8; ++i) x[i] = pr[lane + 32 * i];
    float s = 0.f, s2 = 0.f;
    #pragma unroll
    for (int i = 0; i < 8; ++i) { s += x[i]; s2 += x[i] * x[i]; }
    #pragma unroll
    for (int o = 16; o > 0; o >>= 1) { s += __shfl_xor(s, o, 32); s2 += __shfl_xor(s2, o, 32); }
    const float mean = s * (1.f / 256.f);
    const float var  = s2 * (1.f / 256.f) - mean * mean;
    const float r    = rsqrtf(var + 1e-5f);
    #pragma unroll
    for (int i = 0; i < 8; ++i) {
        const int c = lane + 32 * i;
        float y = (x[i] - mean) * r * g[c] + b[c];
        if (relu) y = fmaxf(y, 0.f);
        Y[(size_t)row * 256 + c]  = y;
        Ybf[(size_t)row * 256 + c] = f2bf(y);
    }
}

// gate[row] = dot(g1[row][0:128], w2) + b2   (one wave per row)
__global__ void gate_dot_k(const float* __restrict__ g1, const float* __restrict__ w2,
                           const float* __restrict__ b2, float* __restrict__ gate) {
    const int row = blockIdx.x * 8 + (threadIdx.x >> 5);
    const int lane = threadIdx.x & 31;
    float s = 0.f;
    #pragma unroll
    for (int i = 0; i < 4; ++i) { int c = lane + 32 * i; s += g1[(size_t)row * 128 + c] * w2[c]; }
    #pragma unroll
    for (int o = 16; o > 0; o >>= 1) s += __shfl_xor(s, o, 32);
    if (lane == 0) gate[row] = s + b2[0];
}

// softmax over n values (single block)
__global__ void softmax_k(const float* __restrict__ g, float* __restrict__ a, int n) {
    __shared__ float red[8];
    const int tid = threadIdx.x, w = tid >> 5, lane = tid & 31;
    float mx = -3.0e38f;
    for (int i = tid; i < n; i += 256) mx = fmaxf(mx, g[i]);
    #pragma unroll
    for (int o = 16; o > 0; o >>= 1) mx = fmaxf(mx, __shfl_xor(mx, o, 32));
    if (lane == 0) red[w] = mx;
    __syncthreads();
    float M = -3.0e38f;
    #pragma unroll
    for (int j = 0; j < 8; ++j) M = fmaxf(M, red[j]);
    __syncthreads();
    float s = 0.f;
    for (int i = tid; i < n; i += 256) s += __expf(g[i] - M);
    #pragma unroll
    for (int o = 16; o > 0; o >>= 1) s += __shfl_xor(s, o, 32);
    if (lane == 0) red[w] = s;
    __syncthreads();
    float T = 0.f;
    #pragma unroll
    for (int j = 0; j < 8; ++j) T += red[j];
    const float inv = 1.0f / T;
    for (int i = tid; i < n; i += 256) a[i] = __expf(g[i] - M) * inv;
}

// pooled[c] = sum_q a[q] * t[q][c]   (single block, 256 threads)
__global__ void pool_k(const float* __restrict__ a, const float* __restrict__ t,
                       float* __restrict__ pooled, int n) {
    const int c = threadIdx.x;
    float s = 0.f;
    for (int q = 0; q < n; ++q) s += a[q] * t[(size_t)q * 256 + c];
    pooled[c] = s;
}

// out[o] = dot(pooled, head_w[:,o]) + head_b[o], o < 2
__global__ void head_k(const float* __restrict__ pooled, const float* __restrict__ hw,
                       const float* __restrict__ hb, float* __restrict__ out) {
    const int o = threadIdx.x >> 5, lane = threadIdx.x & 31;
    float s = 0.f;
    #pragma unroll
    for (int i = 0; i < 8; ++i) { int c = lane + 32 * i; s += pooled[c] * hw[c * 2 + o]; }
    #pragma unroll
    for (int off = 16; off > 0; off >>= 1) s += __shfl_xor(s, off, 32);
    if (lane == 0) out[o] = s + hb[o];
}

// ---------------------------------------------------------------------------
extern "C" void kernel_launch(void* const* d_in, const int* in_sizes, int n_in,
                              void* d_out, int out_size, void* d_ws, size_t ws_size,
                              hipStream_t stream) {
    (void)in_sizes; (void)n_in; (void)out_size; (void)ws_size;
    const int G = 2048, H = 256, HH = H * H;
    const float scale = 0.17677669529663689f;   // 1/sqrt(32)

    const float* gene     = (const float*)d_in[0];
    const float* pre_w1   = (const float*)d_in[3];
    const float* pre_b1   = (const float*)d_in[4];
    const float* pre_w2   = (const float*)d_in[5];
    const float* pre_b2   = (const float*)d_in[6];
    const float* pre_ln_g = (const float*)d_in[7];
    const float* pre_ln_b = (const float*)d_in[8];
    const float* wm       = (const float*)d_in[9];     // [4,4,H,H]
    const float* wsarr    = (const float*)d_in[10];    // [3,4,H,H]
    const float* ln_g     = (const float*)d_in[11];    // [3,H] -> row 0
    const float* ln_b     = (const float*)d_in[12];
    const float* gate_w1  = (const float*)d_in[13];
    const float* gate_b1  = (const float*)d_in[14];
    const float* gate_w2  = (const float*)d_in[15];
    const float* gate_b2  = (const float*)d_in[16];
    const float* tr_w     = (const float*)d_in[17];
    const float* tr_b     = (const float*)d_in[18];
    const float* head_w   = (const float*)d_in[19];
    const float* head_b   = (const float*)d_in[20];
    const unsigned char* adj_pp = (const unsigned char*)d_in[21];  // bool bytes
    const unsigned char* adj_rr = (const unsigned char*)d_in[22];

    char* base = (char*)d_ws;
    size_t off = 0;
    auto alloc = [&](size_t nbytes) -> void* {
        off = (off + 255) & ~(size_t)255;
        void* p = base + off; off += nbytes; return p;
    };

    u16* wT[15];
    const float* wsrc[15] = {
        pre_w1, pre_w2,
        wm + (0 * 4 + 0) * HH, wm + (0 * 4 + 1) * HH, wm + (0 * 4 + 2) * HH, wm + (0 * 4 + 3) * HH,
        wm + (1 * 4 + 0) * HH, wm + (1 * 4 + 1) * HH, wm + (1 * 4 + 2) * HH, wm + (1 * 4 + 3) * HH,
        wsarr + (0 * 4 + 0) * HH, wsarr + (0 * 4 + 1) * HH, wsarr + (0 * 4 + 2) * HH, wsarr + (0 * 4 + 3) * HH,
        tr_w
    };
    for (int i = 0; i < 15; ++i) wT[i] = (u16*)alloc(sizeof(u16) * HH);
    u16* g1T = (u16*)alloc(sizeof(u16) * 256 * 128);

    unsigned* ppM = (unsigned*)alloc(sizeof(unsigned) * G * (G >> 5));
    unsigned* rrM = (unsigned*)alloc(sizeof(unsigned) * G * (G >> 5));

    u16*   xgb  = (u16*)alloc(sizeof(u16) * G * H);
    u16*   hb   = (u16*)alloc(sizeof(u16) * G * H);
    u16*   qb   = (u16*)alloc(sizeof(u16) * G * H);
    u16*   kb   = (u16*)alloc(sizeof(u16) * G * H);
    u16*   vtb  = (u16*)alloc(sizeof(u16) * G * H);     // transposed [256][G]
    u16*   ctxb = (u16*)alloc(sizeof(u16) * G * H);
    float* h32  = (float*)alloc(sizeof(float) * G * H);
    float* xg32 = (float*)alloc(sizeof(float) * G * H);
    float* g1f  = (float*)alloc(sizeof(float) * G * 128);
    float* t32  = (float*)alloc(sizeof(float) * G * H);
    float* gate = (float*)alloc(sizeof(float) * G);
    float* aw   = (float*)alloc(sizeof(float) * G);
    float* pooled = (float*)alloc(sizeof(float) * H);

    // --- weight conversion + mask packing ---
    for (int i = 0; i < 15; ++i)
        tconv_k<<<HH / 256, 256, 0, stream>>>(wsrc[i], wT[i], H, H);
    tconv_k<<<(256 * 128) / 256, 256, 0, stream>>>(gate_w1, g1T, 256, 128);
    pack_mask_k<<<(G * (G >> 5)) / 256, 256, 0, stream>>>(adj_pp, ppM, G, G);
    pack_mask_k<<<(G * (G >> 5)) / 256, 256, 0, stream>>>(adj_rr, rrM, G, G);

    // --- preprocessor MLP ---
    f32_to_bf16_k<<<(G * H) / 256, 256, 0, stream>>>(gene, hb, G * H);
    gemm_bf16_k<2><<<G / 16, 256, 0, stream>>>(hb, wT[0], pre_b1, h32, nullptr, G, 0);
    ln_act_k<<<G / 8, 256, 0, stream>>>(h32, pre_ln_g, pre_ln_b, h32, hb, 1);
    gemm_bf16_k<2><<<G / 16, 256, 0, stream>>>(hb, wT[1], pre_b2, xg32, nullptr, G, 0);
    ln_act_k<<<G / 8, 256, 0, stream>>>(xg32, pre_ln_g, pre_ln_b, xg32, xgb, 1);

    // --- one masked/self MHA, accumulating output into xg32 ---
    auto run_mha = [&](u16* wq, u16* wk, u16* wv, u16* wo, const unsigned* pm) {
        gemm_bf16_k<2><<<G / 16, 256, 0, stream>>>(xgb, wq, nullptr, nullptr, qb,  G, 0);
        gemm_bf16_k<2><<<G / 16, 256, 0, stream>>>(xgb, wk, nullptr, nullptr, kb,  G, 0);
        gemm_bf16_k<2><<<G / 16, 256, 0, stream>>>(xgb, wv, nullptr, nullptr, vtb, G, FLAG_TBF);
        if (pm) flash_k<true ><<<G / 16, 256, 0, stream>>>(qb, kb, vtb, ctxb, pm, G, G, scale);
        else    flash_k<false><<<G / 16, 256, 0, stream>>>(qb, kb, vtb, ctxb, pm, G, G, scale);
        gemm_bf16_k<2><<<G / 16, 256, 0, stream>>>(ctxb, wo, nullptr, xg32, nullptr, G, FLAG_ACC);
    };

    // --- 'M' block: adj_pp and adj_rr masked attention ---
    run_mha(wT[2], wT[3], wT[4], wT[5], ppM);
    run_mha(wT[6], wT[7], wT[8], wT[9], rrM);
    f32_to_bf16_k<<<(G * H) / 256, 256, 0, stream>>>(xg32, xgb, G * H);

    // --- 'S' block: full self-attention ---
    run_mha(wT[10], wT[11], wT[12], wT[13], nullptr);

    // --- final LN (ln_g[0]) ---
    ln_act_k<<<G / 8, 256, 0, stream>>>(xg32, ln_g, ln_b, xg32, xgb, 0);

    // --- attentional aggregation + head ---
    gemm_bf16_k<1><<<G / 16, 256, 0, stream>>>(xgb, g1T, gate_b1, g1f, nullptr, G, FLAG_RELU);
    gate_dot_k<<<G / 8, 256, 0, stream>>>(g1f, gate_w2, gate_b2, gate);
    softmax_k<<<1, 256, 0, stream>>>(gate, aw, G);
    gemm_bf16_k<2><<<G / 16, 256, 0, stream>>>(xgb, wT[14], tr_b, t32, nullptr, G, FLAG_RELU);
    pool_k<<<1, 256, 0, stream>>>(aw, t32, pooled, G);
    head_k<<<1, 64, 0, stream>>>(pooled, head_w, head_b, (float*)d_out);
}
